// swapHead_type_49478023250679
// MI455X (gfx1250) — matrix operand order repaired and checked by reading, not compiled, not run on hardware
//
#include <hip/hip_runtime.h>

typedef __attribute__((ext_vector_type(16))) _Float16 v16h;
typedef __attribute__((ext_vector_type(8)))  float    v8f;

#define NODE_DIM 64   // per-node feature dim
#define HID      256  // hidden width (16 N-tiles of 16)

// ---------------------------------------------------------------------------
// Repack W1 (f32, [128 x 256] row-major) into the WMMA f16 B-operand layout.
// B tile (kt, n) covers K = 32*kt .. 32*kt+31, columns 16*n .. 16*n+15.
// CDNA5 16-bit B operand (wave32): half-lane g of column (lane&15) holds
// K = {g*8+i, 16+g*8+i} for i = 0..7.
// Linear layout: ((kt*16 + n)*32 + lane)*16 + i   -> 32768 halfs = 64 KB.
// ---------------------------------------------------------------------------
__global__ __launch_bounds__(256) void repack_w1_f16(const float* __restrict__ W1,
                                                     _Float16* __restrict__ w1pk)
{
    int idx = blockIdx.x * blockDim.x + threadIdx.x;
    if (idx >= 4 * 16 * 32 * 16) return;
    int i    = idx & 15;
    int lane = (idx >> 4) & 31;
    int n    = (idx >> 9) & 15;
    int kt   = idx >> 13;
    int g    = lane >> 4;
    int c    = lane & 15;
    int K    = kt * 32 + g * 8 + (i & 7) + ((i >> 3) << 4);
    int col  = n * 16 + c;
    w1pk[idx] = (_Float16)W1[K * HID + col];
}

// ---------------------------------------------------------------------------
// Main kernel: one wave handles 32 edges (two 16-row M-tiles) per iteration,
// so each B tile loaded from the packed W1 feeds 4 WMMAs (fwd+rev x 2 tiles).
// ---------------------------------------------------------------------------
__global__ __launch_bounds__(256) void edge_mlp_wmma(
    const float*    __restrict__ x,      // [N, 64]
    const int*      __restrict__ eidx,   // [2, E]
    const float*    __restrict__ b1,     // [256]
    const float*    __restrict__ W2,     // [256]
    const float*    __restrict__ b2,     // [1]
    const _Float16* __restrict__ w1pk,   // repacked W1, 32768 halfs
    float*          __restrict__ out,    // [E]
    int E)
{
    const int lane = threadIdx.x & 31;
    const int g    = lane >> 4;   // half-wave group
    const int m    = lane & 15;   // row-in-tile / column id
    const long nwaves = (long)gridDim.x * (blockDim.x >> 5);
    long gwave = (long)blockIdx.x * (blockDim.x >> 5) + (threadIdx.x >> 5);
    const int numTiles = (E + 31) >> 5;          // 32 edges per tile
    const v16h* __restrict__ bmat = (const v16h*)w1pk;
    const float bias2 = b2[0];

    for (long tile = gwave; tile < numTiles; tile += nwaves) {
        const int ebase = (int)tile * 32;

        // -------- gather 2x16 edges: A tiles (16 x 32 f16), src & dst --------
        v16h As[2][2], Ad[2][2];   // [mtile][ktile]
        #pragma unroll
        for (int mt = 0; mt < 2; ++mt) {
            int e  = ebase + mt * 16 + m;
            int ec = e < E ? e : E - 1;
            int si = eidx[ec];
            int di = eidx[E + ec];
            const float* xs = x + (long)si * NODE_DIM;
            const float* xd = x + (long)di * NODE_DIM;
            #pragma unroll
            for (int t = 0; t < 2; ++t) {
                // ISA 16-bit A layout: half-lane g -> K offsets {g*8+i, 16+g*8+i}
                const int base = 32 * t + g * 8;
                #pragma unroll
                for (int i = 0; i < 8; ++i) {
                    As[mt][t][i]     = (_Float16)xs[base + i];
                    As[mt][t][8 + i] = (_Float16)xs[base + 16 + i];
                    Ad[mt][t][i]     = (_Float16)xd[base + i];
                    Ad[mt][t][8 + i] = (_Float16)xd[base + 16 + i];
                }
            }
        }

        // -------- loop over 16 hidden-unit tiles, fuse ReLU + W2 dot ---------
        float acc0[8] = {0.f, 0.f, 0.f, 0.f, 0.f, 0.f, 0.f, 0.f};
        float acc1[8] = {0.f, 0.f, 0.f, 0.f, 0.f, 0.f, 0.f, 0.f};
        #pragma unroll 2
        for (int n = 0; n < 16; ++n) {
            v16h Bw0 = bmat[(0 * 16 + n) * 32 + lane];  // Wa, K  0..31
            v16h Bw1 = bmat[(1 * 16 + n) * 32 + lane];  // Wa, K 32..63
            v16h Bw2 = bmat[(2 * 16 + n) * 32 + lane];  // Wb, K 64..95
            v16h Bw3 = bmat[(3 * 16 + n) * 32 + lane];  // Wb, K 96..127

            v8f cf0 = {}, cr0 = {}, cf1 = {}, cr1 = {};
            // K 0..31 (Wa) : fwd uses src, rev uses dst
            cf0 = __builtin_amdgcn_wmma_f32_16x16x32_f16(false, As[0][0], false, Bw0, (short)0, cf0, false, false);
            cr0 = __builtin_amdgcn_wmma_f32_16x16x32_f16(false, Ad[0][0], false, Bw0, (short)0, cr0, false, false);
            cf1 = __builtin_amdgcn_wmma_f32_16x16x32_f16(false, As[1][0], false, Bw0, (short)0, cf1, false, false);
            cr1 = __builtin_amdgcn_wmma_f32_16x16x32_f16(false, Ad[1][0], false, Bw0, (short)0, cr1, false, false);
            // K 32..63 (Wa)
            cf0 = __builtin_amdgcn_wmma_f32_16x16x32_f16(false, As[0][1], false, Bw1, (short)0, cf0, false, false);
            cr0 = __builtin_amdgcn_wmma_f32_16x16x32_f16(false, Ad[0][1], false, Bw1, (short)0, cr0, false, false);
            cf1 = __builtin_amdgcn_wmma_f32_16x16x32_f16(false, As[1][1], false, Bw1, (short)0, cf1, false, false);
            cr1 = __builtin_amdgcn_wmma_f32_16x16x32_f16(false, Ad[1][1], false, Bw1, (short)0, cr1, false, false);
            // K 64..95 (Wb) : fwd uses dst, rev uses src
            cf0 = __builtin_amdgcn_wmma_f32_16x16x32_f16(false, Ad[0][0], false, Bw2, (short)0, cf0, false, false);
            cr0 = __builtin_amdgcn_wmma_f32_16x16x32_f16(false, As[0][0], false, Bw2, (short)0, cr0, false, false);
            cf1 = __builtin_amdgcn_wmma_f32_16x16x32_f16(false, Ad[1][0], false, Bw2, (short)0, cf1, false, false);
            cr1 = __builtin_amdgcn_wmma_f32_16x16x32_f16(false, As[1][0], false, Bw2, (short)0, cr1, false, false);
            // K 96..127 (Wb)
            cf0 = __builtin_amdgcn_wmma_f32_16x16x32_f16(false, Ad[0][1], false, Bw3, (short)0, cf0, false, false);
            cr0 = __builtin_amdgcn_wmma_f32_16x16x32_f16(false, As[0][1], false, Bw3, (short)0, cr0, false, false);
            cf1 = __builtin_amdgcn_wmma_f32_16x16x32_f16(false, Ad[1][1], false, Bw3, (short)0, cf1, false, false);
            cr1 = __builtin_amdgcn_wmma_f32_16x16x32_f16(false, As[1][1], false, Bw3, (short)0, cr1, false, false);

            float bb = b1[n * 16 + m];
            float wv = W2[n * 16 + m];
            #pragma unroll
            for (int j = 0; j < 8; ++j) {
                float hf0 = cf0[j] + bb; hf0 = hf0 > 0.f ? hf0 : 0.f;
                float hr0 = cr0[j] + bb; hr0 = hr0 > 0.f ? hr0 : 0.f;
                float hf1 = cf1[j] + bb; hf1 = hf1 > 0.f ? hf1 : 0.f;
                float hr1 = cr1[j] + bb; hr1 = hr1 > 0.f ? hr1 : 0.f;
                acc0[j] += (hf0 + hr0) * wv;
                acc1[j] += (hf1 + hr1) * wv;
            }
        }

        // -------- reduce over the 16 columns held by each half-wave ----------
        #pragma unroll
        for (int j = 0; j < 8; ++j) {
            float v0 = acc0[j];
            v0 += __shfl_xor(v0, 1, 32);
            v0 += __shfl_xor(v0, 2, 32);
            v0 += __shfl_xor(v0, 4, 32);
            v0 += __shfl_xor(v0, 8, 32);
            acc0[j] = v0;
            float v1 = acc1[j];
            v1 += __shfl_xor(v1, 1, 32);
            v1 += __shfl_xor(v1, 2, 32);
            v1 += __shfl_xor(v1, 4, 32);
            v1 += __shfl_xor(v1, 8, 32);
            acc1[j] = v1;
        }

        // lanes 0-15 held rows 0..7 (M=j), lanes 16-31 rows 8..15 (M=j+8)
        if (m == 0) {
            #pragma unroll
            for (int j = 0; j < 8; ++j) {
                int eo0 = ebase + g * 8 + j;
                if (eo0 < E) out[eo0] = 0.5f * acc0[j] + bias2;
                int eo1 = ebase + 16 + g * 8 + j;
                if (eo1 < E) out[eo1] = 0.5f * acc1[j] + bias2;
            }
        }
    }
}

extern "C" void kernel_launch(void* const* d_in, const int* in_sizes, int n_in,
                              void* d_out, int out_size, void* d_ws, size_t ws_size,
                              hipStream_t stream)
{
    (void)n_in; (void)out_size; (void)ws_size;
    const float* x    = (const float*)d_in[0];
    const int*   eidx = (const int*)d_in[1];
    // d_in[2] edge_attr3, d_in[3] edge_attr4, d_in[4] batch : unused by reference
    const float* W1   = (const float*)d_in[5];
    const float* b1   = (const float*)d_in[6];
    const float* W2   = (const float*)d_in[7];
    const float* b2   = (const float*)d_in[8];
    float* out = (float*)d_out;

    int E = in_sizes[1] / 2;
    _Float16* w1pk = (_Float16*)d_ws;   // 64 KB of scratch

    repack_w1_f16<<<128, 256, 0, stream>>>(W1, w1pk);
    edge_mlp_wmma<<<1024, 256, 0, stream>>>(x, eidx, b1, W2, b2, w1pk, out, E);
}
